// SFMCNN_54305566490846
// MI455X (gfx1250) — matrix operand order repaired and checked by reading, not compiled, not run on hardware
//
#include <hip/hip_runtime.h>

typedef __attribute__((ext_vector_type(16))) _Float16 v16h;
typedef __attribute__((ext_vector_type(8)))  _Float16 v8h;
typedef __attribute__((ext_vector_type(8)))  float    v8f;

#define BATCH 128

// ---------------- helpers ----------------

__device__ inline v16h combine16(v8h lo, v8h hi) {
    v16h r;
    #pragma unroll
    for (int j = 0; j < 8; ++j) { r[j] = lo[j]; r[j + 8] = hi[j]; }
    return r;
}

// f32 -> f16 bulk convert
__global__ void cvt_kernel(const float* __restrict__ in, _Float16* __restrict__ out, int n) {
    int i = blockIdx.x * blockDim.x + threadIdx.x;
    if (i < n) out[i] = (_Float16)in[i];
}

// wsq[o] = sum_k w[o,k]^2
__global__ void wsq_kernel(const float* __restrict__ w, int O, int K, float* __restrict__ out) {
    int o = blockIdx.x * blockDim.x + threadIdx.x;
    if (o >= O) return;
    const float* row = w + (size_t)o * K;
    float s = 0.f;
    for (int k = 0; k < K; ++k) { float v = row[k]; s += v * v; }
    out[o] = s;
}

// xsq conv: out[b,y,x] = sum_{c,ky,kx} in[b,c,y+ky-pad,x+kx-pad]^2
__global__ void sqconv_kernel(const float* __restrict__ in, int C, int H, int Wd,
                              int kh, int pad, int bstride, float* __restrict__ out) {
    int idx = blockIdx.x * blockDim.x + threadIdx.x;
    int HW = H * Wd;
    if (idx >= BATCH * HW) return;
    int b = idx / HW; int rem = idx - b * HW; int y = rem / Wd; int x = rem - y * Wd;
    float s = 0.f;
    for (int c = 0; c < C; ++c) {
        const float* ch = in + (size_t)b * bstride + (size_t)c * HW;
        for (int ky = 0; ky < kh; ++ky) {
            int yy = y + ky - pad;
            if (yy < 0 || yy >= H) continue;
            for (int kx = 0; kx < kh; ++kx) {
                int xx = x + kx - pad;
                if (xx < 0 || xx >= Wd) continue;
                float v = ch[yy * Wd + xx];
                s += v * v;
            }
        }
    }
    out[idx] = s;
}

// Layer 1: direct RBF 5x5 conv (C=1) + triangle + crelu + fused 2x2 alpha-pool
__global__ void layer1_kernel(const float* __restrict__ xt,   // x + t*2304, batch stride 3*2304
                              const float* __restrict__ w1,   // (64,25)
                              const float* __restrict__ wsq,  // (64)
                              const float* __restrict__ xsq,  // (B,48,48)
                              const float* __restrict__ twp, const float* __restrict__ cbp,
                              const float* __restrict__ alp,
                              float* __restrict__ h1) {       // (B,64,24,24)
    int idx = blockIdx.x * blockDim.x + threadIdx.x;
    if (idx >= BATCH * 64 * 24 * 24) return;
    int px = idx % 24; int t1 = idx / 24;
    int py = t1 % 24;  int t2 = t1 / 24;
    int o  = t2 % 64;  int b  = t2 / 64;
    float tw = twp[0], cb = cbp[0], al = alp[0];
    float ap0 = al * al * al, ap1 = al * al, ap2 = al, ap3 = 1.f;
    const float* wr = w1 + o * 25;
    const float* xb = xt + (size_t)b * (3 * 48 * 48);
    float wq = wsq[o];
    float acc = 0.f;
    #pragma unroll
    for (int dy = 0; dy < 2; ++dy) {
        #pragma unroll
        for (int dx = 0; dx < 2; ++dx) {
            int oy = 2 * py + dy, ox = 2 * px + dx;
            float cross = 0.f;
            #pragma unroll
            for (int ky = 0; ky < 5; ++ky) {
                int yy = oy + ky - 2;
                if (yy < 0 || yy >= 48) continue;
                #pragma unroll
                for (int kx = 0; kx < 5; ++kx) {
                    int xx = ox + kx - 2;
                    if (xx < 0 || xx >= 48) continue;
                    cross += xb[yy * 48 + xx] * wr[ky * 5 + kx];
                }
            }
            float d = xsq[(size_t)b * 2304 + oy * 48 + ox] - 2.f * cross + wq;
            d = fmaxf(d, 0.f);
            float tri = 1.f - fminf(d, tw) / tw;
            float act = (tri >= cb) ? tri : 0.f;
            float apw = (dy == 0) ? (dx == 0 ? ap0 : ap1) : (dx == 0 ? ap2 : ap3);
            acc += act * apw;
        }
    }
    h1[idx] = acc * 0.25f;
}

// Implicit-GEMM RBF conv (3x3, pad 1) via WMMA f16->f32, LDS-staged B operand.
// Each block owns one group of 16 output pixels; waves cover M-tiles.
// cross[o,p] = sum_k W[o,k]*patch[k,p]; epilogue: xsq - 2*cross + wsq, clamp,
// triangle, crelu.  Output written f32 or f16 (f16 feeds the FC GEMM directly).
template <int M, int C, int H, int WPB, bool OUTH>
__global__ void __launch_bounds__(WPB * 32)
conv_wmma_kernel(const float* __restrict__ in,      // (B,C,H,H) f32
                 const _Float16* __restrict__ wh,   // (M, C*9) f16
                 const float* __restrict__ wsq,     // (M)
                 const float* __restrict__ xsq,     // (B,H,H)
                 const float* __restrict__ twp, const float* __restrict__ cbp,
                 float* __restrict__ actf,          // (B,M,H,H) if !OUTH
                 _Float16* __restrict__ acth) {     // (B,M,H,H) if OUTH
    constexpr int Wd = H, HW = H * H, K = C * 9, mtiles = M / 16;
    __shared__ __align__(32) _Float16 lpatch[16 * K];   // [n][k] patch tile, f16
    int tid = threadIdx.x;
    int wave = tid >> 5, lane = tid & 31;
    int wid = blockIdx.x * WPB + wave;
    int mt = wid % mtiles;
    int nt = wid / mtiles;                 // same for all waves in block
    // ---- cooperative staging of the 16-pixel patch (once per block) ----
    {
        int p0 = ((blockIdx.x * WPB) / mtiles) * 16;
        for (int i = tid; i < 16 * K; i += WPB * 32) {
            int n = i / K, k = i - n * K;
            int p = p0 + n;
            int b = p / HW; int rem = p - b * HW; int y = rem / Wd; int x = rem - y * Wd;
            int c = k / 9; int r = k - c * 9; int ky = r / 3; int kx = r - ky * 3;
            int yy = y + ky - 1, xx = x + kx - 1;
            float v = 0.f;
            if (yy >= 0 && yy < H && xx >= 0 && xx < Wd)
                v = in[(size_t)b * C * HW + (size_t)c * HW + yy * Wd + xx];
            lpatch[i] = (_Float16)v;
        }
    }
    __syncthreads();
    // ---- WMMA main loop ----
    int kgrp = lane >> 4, lm = lane & 15;
    int n = nt * 16 + lm;
    int b = n / HW; int rem = n - b * HW; int y = rem / Wd; int x = rem - y * Wd;
    const _Float16* wrow = wh + (size_t)(mt * 16 + lm) * K;   // A row (f16)
    const _Float16* prow = &lpatch[lm * K];                   // B column (f16, LDS)
    v8f acc = {};
    for (int ks = 0; ks < K / 32; ++ks) {
        int kb = ks * 32;
        v16h a = *reinterpret_cast<const v16h*>(wrow + kb + kgrp * 16);
        v16h bf = *reinterpret_cast<const v16h*>(prow + kb + kgrp * 16);
        acc = __builtin_amdgcn_wmma_f32_16x16x32_f16(false, a, false, bf,
                                                     (short)0, acc, false, false);
    }
    // ---- epilogue ----
    float tw = twp[0], cb = cbp[0];
    float xs = xsq[n];
    #pragma unroll
    for (int v = 0; v < 8; ++v) {
        int o = mt * 16 + kgrp * 8 + v;     // C/D layout: lanes 16-31 hold M=8..15
        float d = xs - 2.f * acc[v] + wsq[o];
        d = fmaxf(d, 0.f);
        float tri = 1.f - fminf(d, tw) / tw;
        float a2 = (tri >= cb) ? tri : 0.f;
        size_t oi = (size_t)b * M * HW + (size_t)o * HW + y * Wd + x;
        if constexpr (OUTH) acth[oi] = (_Float16)a2; else actf[oi] = a2;
    }
}

// 2x2 non-overlapping alpha-weighted mean pool
__global__ void sfm_pool_kernel(const float* __restrict__ in, int C, int H,
                                const float* __restrict__ alp,
                                float* __restrict__ out) {  // (B,C,H/2,H/2)
    int Ho = H >> 1;
    int idx = blockIdx.x * blockDim.x + threadIdx.x;
    if (idx >= BATCH * C * Ho * Ho) return;
    int px = idx % Ho; int t1 = idx / Ho;
    int py = t1 % Ho;  int t2 = t1 / Ho;
    int c  = t2 % C;   int b  = t2 / C;
    float al = alp[0];
    size_t base = (size_t)b * C * H * H + (size_t)c * H * H + (size_t)(2 * py) * H + 2 * px;
    float v = in[base] * (al * al * al) + in[base + 1] * (al * al)
            + in[base + H] * al + in[base + H + 1];
    out[idx] = v * 0.25f;
}

// FC partial GEMM (all-f16 operands): fc_w_slice(112x36864) @ h3^T, K split 32 ways.
// Rows j in [100,112) compute garbage from a clamped (row 99) A pointer; their
// partial slots are never read by fc_reduce_kernel, so no masking is needed in
// the hot loop (keeps the matrix pipe free of cndmask/hazard bubbles).
__global__ void __launch_bounds__(256)
fc_partial_kernel(const _Float16* __restrict__ h3h,   // (B, 36864) f16 features
                  const _Float16* __restrict__ fcwh,  // fcw_h + t*36864, row stride 110592
                  int accumulate,
                  float* __restrict__ part) {         // (32,8,7,256)
    const int Kt = 36864;
    const int STEPS = 36;                             // (36864/32)/32 k-steps per chunk
    int wid  = blockIdx.x * 8 + (threadIdx.x >> 5);
    int lane = threadIdx.x & 31;
    int mt  = wid % 7;
    int nt  = (wid / 7) % 8;
    int kch = wid / 56;
    int kgrp = lane >> 4, lm = lane & 15;
    int j    = mt * 16 + lm;                          // fc output row 0..111
    int jr   = (j < 100) ? j : 99;                    // clamp: avoids OOB only
    int bcol = nt * 16 + lm;                          // batch column
    const _Float16* arow = fcwh + (size_t)jr * 110592;
    const _Float16* brow = h3h + (size_t)bcol * Kt;
    v8f acc = {};
    int kb0 = kch * STEPS * 32;
    for (int s = 0; s < STEPS; ++s) {
        int kb = kb0 + s * 32;
        v16h a = *reinterpret_cast<const v16h*>(arow + kb + kgrp * 16);
        v16h bf = *reinterpret_cast<const v16h*>(brow + kb + kgrp * 16);
        acc = __builtin_amdgcn_wmma_f32_16x16x32_f16(false, a, false, bf,
                                                     (short)0, acc, false, false);
    }
    float* slot = part + ((size_t)(kch * 8 + nt) * 7 + mt) * 256;
    #pragma unroll
    for (int v = 0; v < 8; ++v) {
        int si = (kgrp * 8 + v) * 16 + lm;
        if (accumulate) slot[si] += acc[v]; else slot[si] = acc[v];
    }
}

__global__ void fc_reduce_kernel(const float* __restrict__ part,
                                 const float* __restrict__ fcb,
                                 float* __restrict__ out) {  // (B,100)
    int idx = blockIdx.x * blockDim.x + threadIdx.x;
    if (idx >= BATCH * 100) return;
    int j = idx % 100, b = idx / 100;
    int mt = j >> 4, mi = j & 15;
    int nt = b >> 4, ni = b & 15;
    float s = fcb[j];
    for (int kch = 0; kch < 32; ++kch)
        s += part[((size_t)(kch * 8 + nt) * 7 + mt) * 256 + mi * 16 + ni];
    out[idx] = s;
}

// ---------------- host orchestration ----------------

extern "C" void kernel_launch(void* const* d_in, const int* in_sizes, int n_in,
                              void* d_out, int out_size, void* d_ws, size_t ws_size,
                              hipStream_t stream) {
    (void)in_sizes; (void)n_in; (void)out_size; (void)ws_size;
    const float* x   = (const float*)d_in[0];   // (128,3,48,48)
    const float* W1  = (const float*)d_in[1];   // (3,64,25)
    const float* W2  = (const float*)d_in[2];   // (3,128,576)
    const float* W3  = (const float*)d_in[3];   // (3,256,1152)
    const float* twv = (const float*)d_in[4];   // (3,3)
    const float* cbv = (const float*)d_in[5];   // (3,3)
    const float* alv = (const float*)d_in[6];   // (3,2)
    const float* fcw = (const float*)d_in[7];   // (100,110592)
    const float* fcb = (const float*)d_in[8];   // (100)
    float* out = (float*)d_out;

    float* ws = (float*)d_ws;
    size_t off = 0;
    auto carve = [&](size_t nfloats) { float* p = ws + off; off += nfloats; return p; };
    float*     xsq1 = carve((size_t)128 * 48 * 48);
    float*     wsq1 = carve(64);
    float*     h1   = carve((size_t)128 * 64 * 24 * 24);
    float*     xsq2 = carve((size_t)128 * 24 * 24);
    float*     wsq2 = carve(128);
    float*     act2 = carve((size_t)128 * 128 * 24 * 24);
    float*     h2   = carve((size_t)128 * 128 * 12 * 12);
    float*     xsq3 = carve((size_t)128 * 12 * 12);
    float*     wsq3 = carve(256);
    _Float16*  h3h  = (_Float16*)carve((size_t)128 * 36864 / 2);      // f16 features
    float*     part = carve((size_t)32 * 8 * 7 * 256);
    _Float16*  wh2  = (_Float16*)carve((size_t)128 * 576 / 2);        // f16 W2 tower
    _Float16*  wh3  = (_Float16*)carve((size_t)256 * 1152 / 2);       // f16 W3 tower
    _Float16*  fcwh = (_Float16*)carve((size_t)100 * 110592 / 2);     // f16 fc_w

    // whole fc_w -> f16 once
    cvt_kernel<<<43200, 256, 0, stream>>>(fcw, fcwh, 100 * 110592);

    for (int t = 0; t < 3; ++t) {
        const float* xt  = x  + (size_t)t * 48 * 48;
        const float* W1t = W1 + (size_t)t * 64 * 25;
        const float* W2t = W2 + (size_t)t * 128 * 576;
        const float* W3t = W3 + (size_t)t * 256 * 1152;
        const float* tw  = twv + t * 3;
        const float* cb  = cbv + t * 3;
        const float* al  = alv + t * 2;

        cvt_kernel<<<288, 256, 0, stream>>>(W2t, wh2, 128 * 576);
        cvt_kernel<<<1152, 256, 0, stream>>>(W3t, wh3, 256 * 1152);
        wsq_kernel<<<1, 64, 0, stream>>>(W1t, 64, 25, wsq1);
        wsq_kernel<<<1, 128, 0, stream>>>(W2t, 128, 576, wsq2);
        wsq_kernel<<<1, 256, 0, stream>>>(W3t, 256, 1152, wsq3);

        // layer 1: xsq, fused conv+triangle+crelu+pool -> (B,64,24,24)
        sqconv_kernel<<<1152, 256, 0, stream>>>(xt, 1, 48, 48, 5, 2, 3 * 48 * 48, xsq1);
        layer1_kernel<<<18432, 256, 0, stream>>>(xt, W1t, wsq1, xsq1, tw + 0, cb + 0, al + 0, h1);

        // layer 2: WMMA implicit GEMM (M=128,K=576,N=73728), then pool -> (B,128,12,12)
        sqconv_kernel<<<288, 256, 0, stream>>>(h1, 64, 24, 24, 3, 1, 64 * 576, xsq2);
        conv_wmma_kernel<128, 64, 24, 8, false><<<4608, 256, 0, stream>>>(
            h1, wh2, wsq2, xsq2, tw + 1, cb + 1, act2, nullptr);
        sfm_pool_kernel<<<9216, 256, 0, stream>>>(act2, 128, 24, al + 1, h2);

        // layer 3: WMMA implicit GEMM (M=256,K=1152,N=18432) -> f16 features
        sqconv_kernel<<<72, 256, 0, stream>>>(h2, 128, 12, 12, 3, 1, 128 * 144, xsq3);
        conv_wmma_kernel<256, 128, 12, 16, true><<<1152, 512, 0, stream>>>(
            h2, wh3, wsq3, xsq3, tw + 2, cb + 2, nullptr, h3h);

        // FC contribution of this tower (K-split WMMA, deterministic partials)
        fc_partial_kernel<<<224, 256, 0, stream>>>(h3h, fcwh + (size_t)t * 36864,
                                                   t > 0 ? 1 : 0, part);
    }
    fc_reduce_kernel<<<50, 256, 0, stream>>>(part, fcb, out);
}
